// DiscreteVAE_83416854823279
// MI455X (gfx1250) — matrix operand order repaired and checked by reading, not compiled, not run on hardware
//
#include <hip/hip_runtime.h>
#include <hip/hip_bf16.h>

// ---------- types ----------
typedef __attribute__((ext_vector_type(16))) __bf16 v16bf;
typedef __attribute__((ext_vector_type(8)))  __bf16 v8bf;
typedef __attribute__((ext_vector_type(8)))  float  v8f;
typedef __attribute__((ext_vector_type(4)))  unsigned int v4u;
typedef __attribute__((ext_vector_type(8)))  int    v8i;
typedef __attribute__((ext_vector_type(4)))  int    v4i;

union Frag16 { v16bf v; v8bf h[2]; };

__device__ __forceinline__ __bf16 f2bf(float x) { return (__bf16)x; }

#if defined(__has_builtin)
#if __has_builtin(__builtin_amdgcn_tensor_load_to_lds) && __has_builtin(__builtin_amdgcn_s_wait_tensorcnt)
#define USE_TDM 1
#endif
#endif
#ifndef USE_TDM
#define USE_TDM 0
#endif

#if USE_TDM
// ---------------------------------------------------------------------
// Issue one TDM 2D tile load: bf16 tile (tile_dim0 = 32 elems = 64B per
// row, tile_rows rows) from row-major [rows][row_stride] global into LDS
// at lds_off, with HW padding of +4 DWORDs every 16 DWORDs -> LDS row
// stride of 80 bytes (= my [.][40] padded layout). OOB rows/cols beyond
// (rem_k, rem_rows) are zero-filled by the TDM.
// ---------------------------------------------------------------------
__device__ __forceinline__ void tdm_load_tile(unsigned lds_off,
                                              const __bf16* gtile,
                                              int rem_k, int rem_rows,
                                              int row_stride_elems,
                                              int tile_rows)
{
    unsigned long long ga = (unsigned long long)(uintptr_t)gtile;
    v4u g0;
    g0.x = 1u;                                   // count=1 (valid), user mode
    g0.y = lds_off;                              // lds_addr (bytes)
    g0.z = (unsigned)(ga & 0xffffffffu);         // global_addr[31:0]
    g0.w = (unsigned)((ga >> 32) & 0x1ffffffu)   // global_addr[56:32]
         | (2u << 30);                           // type = 2 (image/D#)
    unsigned td0 = (unsigned)(rem_k    > 0 ? rem_k    : 0);
    unsigned td1 = (unsigned)(rem_rows > 0 ? rem_rows : 0);
    unsigned long long s0 = (unsigned long long)(unsigned)row_stride_elems;
    v8i g1;
    g1[0] = (int)((1u << 16)      // data_size = 1 -> 2 bytes
                | (1u << 20)      // pad_enable
                | (3u << 22)      // pad_interval: every 16 DWORDs
                | (3u << 25));    // pad_amount: 4 DWORDs
    g1[1] = (int)((td0 & 0xffffu) << 16);                          // [47:32]=atomic addr 0, [63:48]=td0.lo
    g1[2] = (int)(((td0 >> 16) & 0xffffu) | ((td1 & 0xffffu) << 16));
    g1[3] = (int)(((td1 >> 16) & 0xffffu) | (32u << 16));          // tile_dim0 = 32
    g1[4] = (int)((unsigned)tile_rows & 0xffffu);                  // tile_dim1; tile_dim2 = 0
    g1[5] = (int)(unsigned)(s0 & 0xffffffffu);                     // dim0 stride lo32
    g1[6] = (int)(unsigned)((s0 >> 32) & 0xffffu);                 // dim0 stride hi16; dim1 stride lo16=0
    g1[7] = 0;
    v4i gz = {0, 0, 0, 0};
#if __clang_major__ >= 23
    v8i gz8 = {0, 0, 0, 0, 0, 0, 0, 0};
    __builtin_amdgcn_tensor_load_to_lds(g0, g1, gz, gz, gz8, 0);
#else
    __builtin_amdgcn_tensor_load_to_lds(g0, g1, gz, gz, 0);
#endif
}
#endif // USE_TDM

// =====================================================================
// GEMM:  C[M,N] = act(A[M,K] @ Bt[N,K]^T + bias[N])
// A, Bt are bf16 in memory (Bt is B pre-transposed, row-major [N][K]).
// Block tile 128x64, K-step 32, double-buffered LDS staged by the TDM
// (wave 0 -> A tile, wave 1 -> B tile), s_wait_tensorcnt pipelining.
// 8 waves (4 row x 2 col), each wave 32x32 = 2x2 WMMA 16x16x32 bf16.
// =====================================================================
template<bool RELU, bool OUTBF>
__global__ __launch_bounds__(256)
void gemm_tdm_kernel(const __bf16* __restrict__ A, const __bf16* __restrict__ Bt,
                     const float* __restrict__ bias, void* __restrict__ Cout,
                     int M, int N, int K)
{
    constexpr int BM = 128, BN = 64, BK = 32;
    __shared__ __align__(16) __bf16 aT[2][BM][BK + 8];  // [m][k], 80B rows
    __shared__ __align__(16) __bf16 bT[2][BN][BK + 8];  // [n][k], 80B rows

    const int tid  = threadIdx.x;
    const int lane = tid & 31;
    const int w    = tid >> 5;
    const int wr   = w & 3;
    const int wcw  = w >> 2;
    const int lm   = lane & 15;
    const int lh   = lane >> 4;

    const int m0 = blockIdx.y * BM;
    const int n0 = blockIdx.x * BN;
    const int nK = (K + BK - 1) / BK;

    v8f acc[2][2] = {};

#if USE_TDM
    // preload K-step 0 into buffer 0
    if (w == 0) tdm_load_tile((unsigned)(uintptr_t)&aT[0][0][0],
                              A + (size_t)m0 * K, K, M - m0, K, BM);
    if (w == 1) tdm_load_tile((unsigned)(uintptr_t)&bT[0][0][0],
                              Bt + (size_t)n0 * K, K, N - n0, K, BN);
#endif

    for (int i = 0; i < nK; ++i) {
        const int buf = i & 1;
#if USE_TDM
        __syncthreads();   // everyone done reading buf^1 from iteration i-1
        if (i + 1 < nK) {
            const int k1 = (i + 1) * BK;
            if (w == 0) tdm_load_tile((unsigned)(uintptr_t)&aT[buf ^ 1][0][0],
                                      A + (size_t)m0 * K + k1, K - k1, M - m0, K, BM);
            if (w == 1) tdm_load_tile((unsigned)(uintptr_t)&bT[buf ^ 1][0][0],
                                      Bt + (size_t)n0 * K + k1, K - k1, N - n0, K, BN);
            if (w == 0 || w == 1)
                __builtin_amdgcn_s_wait_tensorcnt((short)1);   // tile i done, i+1 in flight
        } else {
            if (w == 0 || w == 1)
                __builtin_amdgcn_s_wait_tensorcnt((short)0);   // last tile done
        }
        __syncthreads();   // tile i visible to all waves
#else
        // fallback: cooperative bf16 staging (no conversion needed)
        const int k0 = i * BK;
        __syncthreads();
        #pragma unroll
        for (int idx = tid; idx < BM * BK; idx += 256) {
            int m = idx >> 5, k = idx & 31;
            __bf16 v = (__bf16)0.0f;
            if (m0 + m < M && k0 + k < K) v = A[(size_t)(m0 + m) * K + k0 + k];
            aT[buf][m][k] = v;
        }
        #pragma unroll
        for (int idx = tid; idx < BN * BK; idx += 256) {
            int n = idx >> 5, k = idx & 31;
            __bf16 v = (__bf16)0.0f;
            if (n0 + n < N && k0 + k < K) v = Bt[(size_t)(n0 + n) * K + k0 + k];
            bT[buf][n][k] = v;
        }
        __syncthreads();
#endif
        // ---- fragments: two aligned 16B ds loads each, per ISA layout ----
        Frag16 a[2], b[2];
        #pragma unroll
        for (int s = 0; s < 2; ++s) {
            const __bf16* p = &aT[buf][wr * 32 + s * 16 + lm][lh * 8];
            a[s].h[0] = *(const v8bf*)p;
            a[s].h[1] = *(const v8bf*)(p + 16);
        }
        #pragma unroll
        for (int t = 0; t < 2; ++t) {
            const __bf16* p = &bT[buf][wcw * 32 + t * 16 + lm][lh * 8];
            b[t].h[0] = *(const v8bf*)p;
            b[t].h[1] = *(const v8bf*)(p + 16);
        }
        #pragma unroll
        for (int s = 0; s < 2; ++s)
            #pragma unroll
            for (int t = 0; t < 2; ++t)
                acc[s][t] = __builtin_amdgcn_wmma_f32_16x16x32_bf16(
                    false, a[s].v, false, b[t].v, (short)0, acc[s][t], false, false);
    }

    // ---- epilogue ----
    #pragma unroll
    for (int t = 0; t < 2; ++t) {
        int col = n0 + wcw * 32 + t * 16 + lm;
        if (col >= N) continue;
        float bv = bias[col];
        #pragma unroll
        for (int s = 0; s < 2; ++s) {
            #pragma unroll
            for (int r = 0; r < 8; ++r) {
                int row = m0 + wr * 32 + s * 16 + lh * 8 + r;
                if (row < M) {
                    float v = acc[s][t][r] + bv;
                    if (RELU) v = v > 0.0f ? v : 0.0f;
                    if (OUTBF) ((__bf16*)Cout)[(size_t)row * N + col] = (__bf16)v;
                    else       ((float*)Cout)[(size_t)row * N + col] = v;
                }
            }
        }
    }
}

// =====================================================================
// f32 -> bf16 conversions
// =====================================================================
__global__ __launch_bounds__(256)
void conv_bf16_kernel(const float* __restrict__ src, __bf16* __restrict__ dst, long n)
{
    long i = (long)blockIdx.x * 1024 + threadIdx.x;
    #pragma unroll
    for (int u = 0; u < 4; ++u, i += 256)
        if (i < n) dst[i] = f2bf(src[i]);
}

// W[K][N] f32 -> WT[N][K] bf16
__global__ __launch_bounds__(256)
void convT_bf16_kernel(const float* __restrict__ W, __bf16* __restrict__ WT, int K, int N)
{
    long total = (long)K * N;
    for (long i = (long)blockIdx.x * 256 + threadIdx.x; i < total; i += (long)gridDim.x * 256) {
        int n = (int)(i / K), k = (int)(i % K);
        WT[i] = f2bf(W[(size_t)k * N + n]);
    }
}

// =====================================================================
// cbnorm[j] = sum_k codebook[j][k]^2          (512 x 512)
// =====================================================================
__global__ __launch_bounds__(256)
void cbnorm_kernel(const float* __restrict__ cb, float* __restrict__ out)
{
    int j = blockIdx.x * 256 + threadIdx.x;
    if (j < 512) {
        float s = 0.0f;
        for (int k = 0; k < 512; ++k) { float v = cb[(size_t)j * 512 + k]; s += v * v; }
        out[j] = s;
    }
}

// =====================================================================
// Fused VQ: 32 obs rows per block; WMMA scores = obs @ cb^T, running
// argmin(cbnorm - 2*score); gather codebook rows into latent (f32 out)
// and z_bf (bf16 copy for the decoder GEMM).
// =====================================================================
__global__ __launch_bounds__(256)
void vq_argmin_gather_kernel(const float* __restrict__ obs,       // [R][512]
                             const float* __restrict__ codebook,  // [512][512]
                             const float* __restrict__ cbnorm,    // [512]
                             float* __restrict__ latent,          // [R][512]
                             __bf16* __restrict__ zbf)            // [R][512]
{
    constexpr int RB = 32, CD = 512, NB = 64, KB = 128;
    __shared__ __align__(16) __bf16 aT[RB][CD + 8];
    __shared__ __align__(16) __bf16 bT[NB][KB + 8];
    __shared__ float redV[RB][4];
    __shared__ int   redI[RB][4];
    __shared__ int   codes_lds[RB];

    const int tid  = threadIdx.x;
    const int lane = tid & 31;
    const int w    = tid >> 5;
    const int wr   = w & 1;
    const int wc   = w >> 1;
    const int lm   = lane & 15;
    const int lh   = lane >> 4;
    const size_t r0 = (size_t)blockIdx.x * RB;

    #pragma unroll 4
    for (int idx = tid; idx < RB * CD; idx += 256) {
        int m = idx >> 9, k = idx & 511;
        aT[m][k] = f2bf(obs[(r0 + m) * CD + k]);
    }

    float bestV[8];
    int   bestI[8];
    #pragma unroll
    for (int r = 0; r < 8; ++r) { bestV[r] = 3.4e38f; bestI[r] = 0; }

    for (int nc = 0; nc < CD / NB; ++nc) {
        v8f acc = {};
        for (int kc = 0; kc < CD / KB; ++kc) {
            __syncthreads();
            #pragma unroll 4
            for (int idx = tid; idx < NB * KB; idx += 256) {
                int n = idx >> 7, k = idx & 127;
                bT[n][k] = f2bf(codebook[(size_t)(nc * NB + n) * CD + kc * KB + k]);
            }
            __syncthreads();
            #pragma unroll
            for (int ks = 0; ks < 4; ++ks) {
                Frag16 a, b;
                const __bf16* pa = &aT[wr * 16 + lm][kc * KB + ks * 32 + lh * 8];
                a.h[0] = *(const v8bf*)pa;
                a.h[1] = *(const v8bf*)(pa + 16);
                const __bf16* pb = &bT[wc * 16 + lm][ks * 32 + lh * 8];
                b.h[0] = *(const v8bf*)pb;
                b.h[1] = *(const v8bf*)(pb + 16);
                acc = __builtin_amdgcn_wmma_f32_16x16x32_bf16(
                    false, a.v, false, b.v, (short)0, acc, false, false);
            }
        }
        int col = nc * NB + wc * 16 + lm;
        float cn = cbnorm[col];
        #pragma unroll
        for (int r = 0; r < 8; ++r) {
            float v = cn - 2.0f * acc[r];
            if (v < bestV[r]) { bestV[r] = v; bestI[r] = col; }
        }
    }

    #pragma unroll
    for (int r = 0; r < 8; ++r) {
        float v = bestV[r]; int i = bestI[r];
        #pragma unroll
        for (int mask = 1; mask <= 8; mask <<= 1) {
            float ov = __shfl_xor(v, mask, 32);
            int   oi = __shfl_xor(i, mask, 32);
            if (ov < v || (ov == v && oi < i)) { v = ov; i = oi; }
        }
        bestV[r] = v; bestI[r] = i;
    }
    if (lm == 0) {
        int mbase = wr * 16 + lh * 8;
        #pragma unroll
        for (int r = 0; r < 8; ++r) {
            redV[mbase + r][wc] = bestV[r];
            redI[mbase + r][wc] = bestI[r];
        }
    }
    __syncthreads();
    if (tid < RB) {
        float v = redV[tid][0]; int i = redI[tid][0];
        #pragma unroll
        for (int c = 1; c < 4; ++c) {
            float ov = redV[tid][c]; int oi = redI[tid][c];
            if (ov < v || (ov == v && oi < i)) { v = ov; i = oi; }
        }
        codes_lds[tid] = i;
    }
    __syncthreads();

    #pragma unroll 4
    for (int idx = tid; idx < RB * CD; idx += 256) {
        int m = idx >> 9, k = idx & 511;
        float v = codebook[(size_t)codes_lds[m] * CD + k];
        latent[(r0 + m) * CD + k] = v;
        zbf[(r0 + m) * CD + k]    = f2bf(v);
    }
}

// =====================================================================
// launch
// =====================================================================
extern "C" void kernel_launch(void* const* d_in, const int* in_sizes, int n_in,
                              void* d_out, int out_size, void* d_ws, size_t ws_size,
                              hipStream_t stream)
{
    (void)in_sizes; (void)n_in; (void)out_size; (void)ws_size;
    const float* x   = (const float*)d_in[0];
    const float* ew1 = (const float*)d_in[1];
    const float* eb1 = (const float*)d_in[2];
    const float* ew2 = (const float*)d_in[3];
    const float* eb2 = (const float*)d_in[4];
    const float* ew3 = (const float*)d_in[5];
    const float* eb3 = (const float*)d_in[6];
    const float* cb  = (const float*)d_in[7];
    const float* dw1 = (const float*)d_in[8];
    const float* db1 = (const float*)d_in[9];
    const float* dw2 = (const float*)d_in[10];
    const float* db2 = (const float*)d_in[11];

    const int B = 8192, D = 784, H = 500, C = 512, S = 16;
    const int CS = C * S;  // 8192

    float* out        = (float*)d_out;
    float* out_recon  = out;
    float* out_obs    = out + (size_t)B * D;
    float* out_latent = out_obs + (size_t)B * CS;

    // ---- workspace carve-up (256B aligned chunks) ----
    char* p = (char*)d_ws;
    auto take = [&](size_t bytes) { char* r = p; p += (bytes + 255) & ~(size_t)255; return r; };
    float*  cbn   = (float*) take(512 * sizeof(float));
    __bf16* xbf   = (__bf16*)take((size_t)B * D  * 2);
    __bf16* h1bf  = (__bf16*)take((size_t)B * H  * 2);
    __bf16* h2bf  = (__bf16*)take((size_t)B * H  * 2);
    __bf16* dhbf  = (__bf16*)take((size_t)B * H  * 2);
    __bf16* zbf   = (__bf16*)take((size_t)B * CS * 2);
    __bf16* ew1T  = (__bf16*)take((size_t)D * H  * 2);   // [H][D]
    __bf16* ew2T  = (__bf16*)take((size_t)H * H  * 2);   // [H][H]
    __bf16* ew3T  = (__bf16*)take((size_t)H * CS * 2);   // [CS][H]
    __bf16* dw1T  = (__bf16*)take((size_t)CS * H * 2);   // [H][CS]
    __bf16* dw2T  = (__bf16*)take((size_t)H * D  * 2);   // [D][H]

    dim3 blk(256);

    cbnorm_kernel<<<dim3(2), blk, 0, stream>>>(cb, cbn);
    { long n = (long)B * D;
      conv_bf16_kernel<<<dim3((unsigned)((n + 1023) / 1024)), blk, 0, stream>>>(x, xbf, n); }
    convT_bf16_kernel<<<dim3(512), blk, 0, stream>>>(ew1, ew1T, D,  H);
    convT_bf16_kernel<<<dim3(512), blk, 0, stream>>>(ew2, ew2T, H,  H);
    convT_bf16_kernel<<<dim3(2048), blk, 0, stream>>>(ew3, ew3T, H,  CS);
    convT_bf16_kernel<<<dim3(2048), blk, 0, stream>>>(dw1, dw1T, CS, H);
    convT_bf16_kernel<<<dim3(512), blk, 0, stream>>>(dw2, dw2T, H,  D);

    // h1 = relu(x @ ew1 + eb1)          [8192,784]x[784,500] -> bf16
    gemm_tdm_kernel<true, true ><<<dim3((H + 63) / 64, B / 128), blk, 0, stream>>>(xbf,  ew1T, eb1, h1bf, B, H, D);
    // h2 = relu(h1 @ ew2 + eb2)         [8192,500]x[500,500] -> bf16
    gemm_tdm_kernel<true, true ><<<dim3((H + 63) / 64, B / 128), blk, 0, stream>>>(h1bf, ew2T, eb2, h2bf, B, H, H);
    // obs = h2 @ ew3 + eb3              [8192,500]x[500,8192] -> f32 (d_out)
    gemm_tdm_kernel<false, false><<<dim3(CS / 64, B / 128), blk, 0, stream>>>(h2bf, ew3T, eb3, out_obs, B, CS, H);

    // VQ argmin + gather (latent f32 + z bf16)
    vq_argmin_gather_kernel<<<dim3(B * S / 32), blk, 0, stream>>>(out_obs, cb, cbn, out_latent, zbf);

    // dh = relu(z @ dw1 + db1)          [8192,8192]x[8192,500] -> bf16
    gemm_tdm_kernel<true, true ><<<dim3((H + 63) / 64, B / 128), blk, 0, stream>>>(zbf, dw1T, db1, dhbf, B, H, CS);
    // recon = dh @ dw2 + db2            [8192,500]x[500,784] -> f32 (d_out)
    gemm_tdm_kernel<false, false><<<dim3((D + 63) / 64, B / 128), blk, 0, stream>>>(dhbf, dw2T, db2, out_recon, B, D, H);
}
